// TransformerEncoderLayer_3908420239574
// MI455X (gfx1250) — compile-verified, matrix operand order checked by reading, not run on hardware
//
#include <hip/hip_runtime.h>
#include <hip/hip_bf16.h>

typedef __bf16 bf16;
typedef __attribute__((ext_vector_type(16))) __bf16 bf16x16;
typedef __attribute__((ext_vector_type(8)))  float  f32x8;
typedef __attribute__((ext_vector_type(4)))  float  f32x4;
typedef __attribute__((ext_vector_type(4)))  unsigned int u32x4;
typedef __attribute__((ext_vector_type(8)))  unsigned int u32x8;

union Frag { bf16x16 v; u32x4 q[2]; };

constexpr int cB = 4, cS = 1024, cD = 1024, cH = 16, cDK = 64, cF = 4096;

__device__ __forceinline__ f32x8 wmma_bf16(bf16x16 a, bf16x16 b, f32x8 c) {
  // D = A(16x32 bf16) x B(32x16 bf16) + C(16x16 f32)
  return __builtin_amdgcn_wmma_f32_16x16x32_bf16(false, a, false, b, (short)0, c,
                                                 false, false);
}

// ---------------------------------------------------------------------------
// Tensor Data Mover: 2-D tile load, global -> LDS, with per-row LDS padding.
// D# per CDNA5 ISA:
//   group0: count=1 | lds_addr | global_addr[56:0] | type=2
//   group1: data_size=2B, pad_enable, pad_interval/amount, dims/strides
// TW = tile width (elements, dim0, contiguous), TH = tile height (rows).
// PADI: pad after 2<<PADI DWORDs; PADA: pad amount = (PADA+1) DWORDs.
// gstride = global row stride in elements.
// ---------------------------------------------------------------------------
template<int TW, int TH, int PADI, int PADA>
__device__ __forceinline__ void tdm_load(const void* lds_ptr, const void* gptr,
                                         int gstride)
{
  unsigned long long ga = (unsigned long long)gptr;
  unsigned ldsa = (unsigned)(unsigned long long)lds_ptr;   // low 32 = LDS offset
  u32x4 g0;
  g0.x = 1u;                                               // count=1 (user D#)
  g0.y = ldsa;                                             // lds_addr [63:32]
  g0.z = (unsigned)ga;                                     // global_addr[31:0]
  g0.w = (unsigned)((ga >> 32) & 0x01ffffffull) | (2u << 30); // addr[56:32]|type=2
  u32x8 g1;
  g1[0] = (1u << 16)                      // data_size = 2 bytes
        | (1u << 20)                      // pad_enable
        | ((unsigned)PADI << 22)
        | ((unsigned)PADA << 25);
  g1[1] = ((unsigned)TW) << 16;                             // tensor_dim0 lo16
  g1[2] = (((unsigned)TW) >> 16) | (((unsigned)TH) << 16);  // dim0 hi | dim1 lo
  g1[3] = (((unsigned)TH) >> 16) | (((unsigned)TW) << 16);  // dim1 hi | tile_dim0
  g1[4] = (unsigned)TH;                                     // tile_dim1, tile_dim2=0
  g1[5] = (unsigned)gstride;                                // tensor_dim0_stride lo32
  g1[6] = 0u;
  g1[7] = 0u;
  asm volatile("tensor_load_to_lds %0, %1" :: "s"(g0), "s"(g1) : "memory");
}

// ---------------------------------------------------------------------------
// Generic bf16 GEMM with TDM double-buffered staging.
// Block tile 128(M) x 64(N), K-step 32. 8 waves = 4(M) x 2(N), wave 32x32.
// A: bf16 [M,K] row-major.  Bt: bf16 [N,K] (pre-transposed weights).
// ---------------------------------------------------------------------------
template<bool RELU, bool RES, bool OUT32, bool OUT16>
__device__ void gemm_body(const bf16* A, int lda,
                          const bf16* Bt, int ldbt,
                          const float* bias,
                          const float* res, int ldr,
                          float* out32, bf16* out16, int ldo,
                          int m0, int n0, int K)
{
  constexpr int LS = 40;                 // LDS row pitch: 32 elems + 16B pad
  __shared__ bf16 Asm[2][128 * LS];
  __shared__ bf16 Bsm[2][64 * LS];
  const int tid  = threadIdx.x;
  const int lane = tid & 31, wave = tid >> 5;
  const int hx   = lane >> 4, l16 = lane & 15;
  const int wm = wave & 3, wn = wave >> 2;

  f32x8 acc[2][2];
#pragma unroll
  for (int i = 0; i < 2; ++i)
#pragma unroll
    for (int j = 0; j < 2; ++j)
#pragma unroll
      for (int e = 0; e < 8; ++e) acc[i][j][e] = 0.f;

  const bf16* Abase = A + (size_t)m0 * lda;
  const bf16* Bbase = Bt + (size_t)n0 * ldbt;
  const int nIter = K / 32;

  if (wave == 0) {                        // prologue: fill buffer 0
    tdm_load<32, 128, 3, 3>(&Asm[0][0], Abase, lda);
    tdm_load<32, 64, 3, 3>(&Bsm[0][0], Bbase, ldbt);
  }

  for (int it = 0; it < nIter; ++it) {
    const int cur = it & 1;
    if (wave == 0) {
      if (it + 1 < nIter) {               // prefetch next tile into other buffer
        tdm_load<32, 128, 3, 3>(&Asm[cur ^ 1][0], Abase + (it + 1) * 32, lda);
        tdm_load<32, 64, 3, 3>(&Bsm[cur ^ 1][0], Bbase + (it + 1) * 32, ldbt);
        __builtin_amdgcn_s_wait_tensorcnt(2);   // oldest pair (cur) complete
      } else {
        __builtin_amdgcn_s_wait_tensorcnt(0);
      }
    }
    __syncthreads();                      // cur buffer ready for everyone

    Frag a[2], b[2];
#pragma unroll
    for (int tm = 0; tm < 2; ++tm) {
      const bf16* p = &Asm[cur][(wm * 32 + tm * 16 + l16) * LS];
      a[tm].q[0] = *(const u32x4*)(p + hx * 8);
      a[tm].q[1] = *(const u32x4*)(p + 16 + hx * 8);
    }
#pragma unroll
    for (int tn = 0; tn < 2; ++tn) {
      const bf16* p = &Bsm[cur][(wn * 32 + tn * 16 + l16) * LS];
      b[tn].q[0] = *(const u32x4*)(p + hx * 8);
      b[tn].q[1] = *(const u32x4*)(p + 16 + hx * 8);
    }
#pragma unroll
    for (int tm = 0; tm < 2; ++tm)
#pragma unroll
      for (int tn = 0; tn < 2; ++tn)
        acc[tm][tn] = wmma_bf16(a[tm].v, b[tn].v, acc[tm][tn]);
    __syncthreads();                      // done reading cur before it is refilled
  }

  // epilogue: C-layout row = j + 8*half, col = lane&15
#pragma unroll
  for (int tm = 0; tm < 2; ++tm)
#pragma unroll
    for (int tn = 0; tn < 2; ++tn) {
      int col = n0 + wn * 32 + tn * 16 + l16;
      float bb = bias[col];
#pragma unroll
      for (int j = 0; j < 8; ++j) {
        int row = m0 + wm * 32 + tm * 16 + hx * 8 + j;
        float vv = acc[tm][tn][j] + bb;
        if (RES)  vv += res[(size_t)row * ldr + col];
        if (RELU) vv = fmaxf(vv, 0.f);
        if (OUT32) out32[(size_t)row * ldo + col] = vv;
        if (OUT16) out16[(size_t)row * ldo + col] = (bf16)vv;
      }
    }
}

// ---------------------------------------------------------------------------
// Prologue conversions: x -> bf16; weights -> bf16 transposed [N,K].
// ---------------------------------------------------------------------------
__global__ void __launch_bounds__(256)
k_cvt_x(const float* in, bf16* out)
{
  int i = (blockIdx.x * 256 + threadIdx.x) * 4;
  f32x4 d = *(const f32x4*)(in + i);
  out[i + 0] = (bf16)d.x; out[i + 1] = (bf16)d.y;
  out[i + 2] = (bf16)d.z; out[i + 3] = (bf16)d.w;
}

__global__ void __launch_bounds__(256)
k_wtrans(const float* in, bf16* out, int K, int N)
{
  __shared__ float tile[32][33];
  const float* src = in + (size_t)blockIdx.z * K * N;
  bf16* dst = out + (size_t)blockIdx.z * K * N;
  const int kb = blockIdx.x * 32, nb = blockIdx.y * 32;
  const int tx = threadIdx.x & 31, ty = threadIdx.x >> 5;
#pragma unroll
  for (int i = 0; i < 32; i += 8)
    tile[ty + i][tx] = src[(size_t)(kb + ty + i) * N + nb + tx];
  __syncthreads();
#pragma unroll
  for (int i = 0; i < 32; i += 8)
    dst[(size_t)(nb + ty + i) * K + kb + tx] = (bf16)tile[tx][ty + i];
}

// ---------------------------------------------------------------------------
// GEMM wrapper kernels
// ---------------------------------------------------------------------------
__global__ void __launch_bounds__(256)
k_qkv(const bf16* xb, const bf16* WTqkv, const float* bq, const float* bk,
      const float* bv, bf16* qkv)
{
  int z  = blockIdx.y;
  int t  = z / (cB * cH);
  int bh = z - t * (cB * cH);
  int b  = bh / cH, h = bh - b * cH;
  const float* bi = (t == 0) ? bq : (t == 1) ? bk : bv;
  gemm_body<false, false, false, true>(
      xb + (size_t)b * cS * cD, cD,
      WTqkv + ((size_t)t * cH + h) * (size_t)cDK * cD, cD,
      bi + h * cDK, nullptr, 0,
      nullptr, qkv + ((size_t)t * (cB * cH) + bh) * cS * cDK, cDK,
      blockIdx.x * 128, 0, cD);
}

__global__ void __launch_bounds__(256)
k_wo(const bf16* ctx, const bf16* WTo, const float* bo, const float* x, float* t1)
{
  gemm_body<false, true, true, false>(
      ctx, cD, WTo, cD, bo, x, cD, t1, nullptr, cD,
      blockIdx.x * 128, blockIdx.y * 64, cD);
}

__global__ void __launch_bounds__(256)
k_ffn1(const bf16* h1b, const bf16* WT1, const float* b1, bf16* a1)
{
  gemm_body<true, false, false, true>(
      h1b, cD, WT1, cD, b1, nullptr, 0, nullptr, a1, cF,
      blockIdx.x * 128, blockIdx.y * 64, cD);
}

__global__ void __launch_bounds__(256)
k_ffn2(const bf16* a1, const bf16* WT2, const float* b2, const float* h1f, float* t2)
{
  gemm_body<false, true, true, false>(
      a1, cF, WT2, cF, b2, h1f, cD, t2, nullptr, cD,
      blockIdx.x * 128, blockIdx.y * 64, cF);
}

// ---------------------------------------------------------------------------
// Flash attention: workgroup = 128 query rows of one (b,h); 8 waves x 16 rows.
// K-block staged by TDM (wave 0), V-block transposed by VALU, P repacked
// through wave-private LDS (same-wave LDS ops are in-order).
// ---------------------------------------------------------------------------
__global__ void __launch_bounds__(256)
k_attn(const bf16* q, const bf16* k, const bf16* v, bf16* ctx)
{
  const int bh = blockIdx.y;
  const int b  = bh / cH, h = bh - b * cH;
  const int tid = threadIdx.x, lane = tid & 31, wave = tid >> 5;
  const int hx = lane >> 4, l16 = lane & 15;
  const int q0 = blockIdx.x * 128 + wave * 16;

  const bf16* qp = q + (size_t)bh * cS * cDK;
  const bf16* kp = k + (size_t)bh * cS * cDK;
  const bf16* vp = v + (size_t)bh * cS * cDK;

  __shared__ bf16 Ks[32 * 72];        // K block, [key][dk], pitch 72 (TDM pad)
  __shared__ bf16 Vs[64 * 40];        // V block transposed, [dk][key], pitch 40
  __shared__ bf16 Ps[8 * 16 * 40];    // per-wave P repack buffer

  Frag aq[2];
  {
    const bf16* p = qp + (size_t)(q0 + l16) * cDK;
#pragma unroll
    for (int c = 0; c < 2; ++c) {
      aq[c].q[0] = *(const u32x4*)(p + c * 32 + hx * 8);
      aq[c].q[1] = *(const u32x4*)(p + c * 32 + 16 + hx * 8);
    }
  }

  float mrow[8], lrow[8];
  f32x8 acc[4];
#pragma unroll
  for (int j = 0; j < 8; ++j) { mrow[j] = -1e30f; lrow[j] = 0.f; }
#pragma unroll
  for (int nt = 0; nt < 4; ++nt)
#pragma unroll
    for (int j = 0; j < 8; ++j) acc[nt][j] = 0.f;

  for (int it = 0; it < cS / 32; ++it) {
    const int key0 = it * 32;
    if (wave == 0)                       // DMA K block: 32 rows x 64 bf16
      tdm_load<64, 32, 4, 3>(&Ks[0], kp + (size_t)key0 * cDK, cDK);
    {                                    // stage V transposed (VALU path)
      int flat = tid * 8;
      int key = flat >> 6, dk = flat & 63;
      u32x4 dV = *(const u32x4*)(vp + (size_t)(key0 + key) * cDK + dk);
      __align__(16) bf16 tmp[8];
      *(u32x4*)tmp = dV;
#pragma unroll
      for (int i = 0; i < 8; ++i) Vs[(dk + i) * 40 + key] = tmp[i];
    }
    if (wave == 0) __builtin_amdgcn_s_wait_tensorcnt(0);
    __syncthreads();

    // ---- scores = Q . K^T ----
    f32x8 sc[2];
#pragma unroll
    for (int nt = 0; nt < 2; ++nt) {
      Frag bk0, bk1;
      const bf16* pk2 = &Ks[(nt * 16 + l16) * 72];
      bk0.q[0] = *(const u32x4*)(pk2 + hx * 8);
      bk0.q[1] = *(const u32x4*)(pk2 + 16 + hx * 8);
      bk1.q[0] = *(const u32x4*)(pk2 + 32 + hx * 8);
      bk1.q[1] = *(const u32x4*)(pk2 + 48 + hx * 8);
      f32x8 cz;
#pragma unroll
      for (int j = 0; j < 8; ++j) cz[j] = 0.f;
      cz = wmma_bf16(aq[0].v, bk0.v, cz);
      cz = wmma_bf16(aq[1].v, bk1.v, cz);
      sc[nt] = cz;
    }

    // ---- online softmax ----
#pragma unroll
    for (int j = 0; j < 8; ++j) {
      float s0 = sc[0][j] * 0.125f;
      float s1 = sc[1][j] * 0.125f;
      float mb = fmaxf(s0, s1);
#pragma unroll
      for (int msk = 8; msk >= 1; msk >>= 1)
        mb = fmaxf(mb, __shfl_xor(mb, msk, 32));
      float mn    = fmaxf(mrow[j], mb);
      float alpha = __expf(mrow[j] - mn);
      mrow[j] = mn;
      float p0 = __expf(s0 - mn);
      float p1 = __expf(s1 - mn);
      float rs = p0 + p1;
#pragma unroll
      for (int msk = 8; msk >= 1; msk >>= 1)
        rs += __shfl_xor(rs, msk, 32);
      lrow[j] = lrow[j] * alpha + rs;
#pragma unroll
      for (int nt = 0; nt < 4; ++nt) acc[nt][j] *= alpha;
      sc[0][j] = p0;
      sc[1][j] = p1;
    }

    // ---- repack P (C-layout -> A-layout) via wave-private LDS ----
    bf16* pw = &Ps[wave * 16 * 40];
#pragma unroll
    for (int nt = 0; nt < 2; ++nt)
#pragma unroll
      for (int j = 0; j < 8; ++j)
        pw[(hx * 8 + j) * 40 + nt * 16 + l16] = (bf16)sc[nt][j];
    Frag ap;
    {
      const bf16* pr = &Ps[wave * 16 * 40 + l16 * 40];
      ap.q[0] = *(const u32x4*)(pr + hx * 8);
      ap.q[1] = *(const u32x4*)(pr + 16 + hx * 8);
    }

    // ---- ctx += P . V ----
#pragma unroll
    for (int nt = 0; nt < 4; ++nt) {
      Frag bv;
      const bf16* pv = &Vs[(nt * 16 + l16) * 40];
      bv.q[0] = *(const u32x4*)(pv + hx * 8);
      bv.q[1] = *(const u32x4*)(pv + 16 + hx * 8);
      acc[nt] = wmma_bf16(ap.v, bv.v, acc[nt]);
    }
    __syncthreads();
  }

  // ---- normalize, store head-concatenated [B,S,D] ----
#pragma unroll
  for (int j = 0; j < 8; ++j) {
    float inv = 1.f / lrow[j];
    int row = q0 + hx * 8 + j;
#pragma unroll
    for (int nt = 0; nt < 4; ++nt) {
      int dk = nt * 16 + l16;
      ctx[((size_t)b * cS + row) * cD + h * cDK + dk] = (bf16)(acc[nt][j] * inv);
    }
  }
}

// ---------------------------------------------------------------------------
// LayerNorm over D=1024, block per row; optional bf16 copy for next GEMM.
// ---------------------------------------------------------------------------
template<bool O16>
__global__ void __launch_bounds__(256)
k_ln(const float* in, const float* g, const float* be, float* o32, bf16* o16)
{
  const int row = blockIdx.x;
  const float* p = in + (size_t)row * cD;
  float s = 0.f, s2 = 0.f;
  for (int i = threadIdx.x; i < cD; i += 256) {
    float vv = p[i]; s += vv; s2 += vv * vv;
  }
#pragma unroll
  for (int m = 16; m >= 1; m >>= 1) {
    s  += __shfl_xor(s,  m, 32);
    s2 += __shfl_xor(s2, m, 32);
  }
  __shared__ float rs[8], rq[8];
  if ((threadIdx.x & 31) == 0) { rs[threadIdx.x >> 5] = s; rq[threadIdx.x >> 5] = s2; }
  __syncthreads();
  float ts = 0.f, tq = 0.f;
#pragma unroll
  for (int i = 0; i < 8; ++i) { ts += rs[i]; tq += rq[i]; }
  const float mu  = ts * (1.f / cD);
  const float var = tq * (1.f / cD) - mu * mu;
  const float inv = rsqrtf(var + 1e-6f);
  for (int i = threadIdx.x; i < cD; i += 256) {
    float vv = (p[i] - mu) * inv * g[i] + be[i];
    o32[(size_t)row * cD + i] = vv;
    if (O16) o16[(size_t)row * cD + i] = (bf16)vv;
  }
}

// ---------------------------------------------------------------------------
extern "C" void kernel_launch(void* const* d_in, const int* in_sizes, int n_in,
                              void* d_out, int out_size, void* d_ws, size_t ws_size,
                              hipStream_t stream)
{
  (void)in_sizes; (void)n_in; (void)out_size; (void)ws_size;
  const float* x   = (const float*)d_in[0];
  const float* Wq  = (const float*)d_in[1];
  const float* bq  = (const float*)d_in[2];
  const float* Wk  = (const float*)d_in[3];
  const float* bk  = (const float*)d_in[4];
  const float* Wv  = (const float*)d_in[5];
  const float* bv  = (const float*)d_in[6];
  const float* Wo  = (const float*)d_in[7];
  const float* bo  = (const float*)d_in[8];
  const float* W1  = (const float*)d_in[9];
  const float* b1  = (const float*)d_in[10];
  const float* W2  = (const float*)d_in[11];
  const float* b2  = (const float*)d_in[12];
  const float* g1  = (const float*)d_in[13];
  const float* be1 = (const float*)d_in[14];
  const float* g2  = (const float*)d_in[15];
  const float* be2 = (const float*)d_in[16];
  float* out = (float*)d_out;

  // workspace layout (128 MB, with reuse)
  char* w = (char*)d_ws;
  const size_t MB = 1024ull * 1024ull;
  bf16*  qkv   = (bf16*)(w);             // q,k,v: 3 x 8 MB bf16
  bf16*  ctxb  = (bf16*)(w + 24 * MB);   // 8 MB (reused as h1 bf16)
  float* t1    = (float*)(w + 32 * MB);  // 16 MB (reused as t2)
  float* h1f   = (float*)(w + 48 * MB);  // 16 MB
  bf16*  a1    = (bf16*)(w + 64 * MB);   // 32 MB
  bf16*  xb    = (bf16*)(w + 96 * MB);   // 8 MB
  bf16*  WTqkv = (bf16*)(w + 104 * MB);  // 6 MB  (q,k,v each [H][DK][D])
  bf16*  WTo   = (bf16*)(w + 110 * MB);  // 2 MB  [D][D]
  bf16*  WT1   = (bf16*)(w + 112 * MB);  // 8 MB  [F][D]
  bf16*  WT2   = (bf16*)(w + 120 * MB);  // 8 MB  [D][F]
  bf16*  h1b   = ctxb;
  float* t2    = t1;

  const size_t QSZ = (size_t)cB * cH * cS * cDK;          // q/k/v elems
  const size_t WH  = (size_t)cH * cD * cDK;               // per-tensor W elems

  // one-time conversions (weights stay L2-resident afterwards: 24 MB bf16)
  k_cvt_x <<<dim3(4096),        256, 0, stream>>>(x, xb);
  k_wtrans<<<dim3(32, 2, 16),   256, 0, stream>>>(Wq, WTqkv,          cD, cDK);
  k_wtrans<<<dim3(32, 2, 16),   256, 0, stream>>>(Wk, WTqkv + WH,     cD, cDK);
  k_wtrans<<<dim3(32, 2, 16),   256, 0, stream>>>(Wv, WTqkv + 2 * WH, cD, cDK);
  k_wtrans<<<dim3(32, 32, 1),   256, 0, stream>>>(Wo, WTo, cD, cD);
  k_wtrans<<<dim3(32, 128, 1),  256, 0, stream>>>(W1, WT1, cD, cF);
  k_wtrans<<<dim3(128, 32, 1),  256, 0, stream>>>(W2, WT2, cF, cD);

  k_qkv       <<<dim3(8, 192), 256, 0, stream>>>(xb, WTqkv, bq, bk, bv, qkv);
  k_attn      <<<dim3(8, 64),  256, 0, stream>>>(qkv, qkv + QSZ, qkv + 2 * QSZ, ctxb);
  k_wo        <<<dim3(32, 16), 256, 0, stream>>>(ctxb, WTo, bo, x, t1);
  k_ln<true>  <<<dim3(4096),   256, 0, stream>>>(t1, g1, be1, h1f, h1b);
  k_ffn1      <<<dim3(32, 64), 256, 0, stream>>>(h1b, WT1, b1, a1);
  k_ffn2      <<<dim3(32, 16), 256, 0, stream>>>(a1, WT2, b2, h1f, t2);
  k_ln<false> <<<dim3(4096),   256, 0, stream>>>(t2, g2, be2, out, nullptr);
}
